// QuantumFeedForward_65481071396749
// MI455X (gfx1250) — matrix operand order repaired and checked by reading, not compiled, not run on hardware
//
#include <hip/hip_runtime.h>
#include <hip/hip_bf16.h>

typedef __attribute__((ext_vector_type(16))) _Float16 v16h;
typedef __attribute__((ext_vector_type(8)))  _Float16 v8h;
typedef __attribute__((ext_vector_type(4)))  _Float16 v4h;
typedef __attribute__((ext_vector_type(8)))  float    v8f;

#define QFF_EMBED  512
#define QFF_FFN    2048
#define QFF_TOK    64     // tokens per block
#define HA_PITCH   40     // halves per LDS row (80B) -> conflict-free b128
#define KCHUNK     32

// ---------------------------------------------------------------------------
// Async global->LDS copy (CDNA5): 16B per lane, tracked by ASYNCcnt.
// ---------------------------------------------------------------------------
__device__ __forceinline__ void async_b128(const _Float16* g, const _Float16* l) {
  const unsigned loff = (unsigned)(uintptr_t)l;   // LDS offset = low 32 bits
  asm volatile("global_load_async_to_lds_b128 %0, %1, off"
               :: "v"(loff), "v"(g) : "memory");
}
#define ASYNC_WAIT(n) asm volatile("s_wait_asynccnt " #n ::: "memory")

// ---------------------------------------------------------------------------
// Kernel A: build the four 16x16 quadratic-form matrices M_w from params.
// expz_w(token) = m^T M_w m with the (-i)^popcount phases folded into U.
// ---------------------------------------------------------------------------
__global__ void qff_build_M(const float* __restrict__ params,
                            float* __restrict__ wsM) {
  __shared__ float Ur[16][16], Ui[16][16];
  const int tid = threadIdx.x;
  const int i = tid >> 4, j = tid & 15;

  Ur[i][j] = (i == j) ? 1.f : 0.f;
  Ui[i][j] = 0.f;
  __syncthreads();

  for (int layer = 0; layer < 2; ++layer) {
    float rr[4][2][2], ri[4][2][2];
    #pragma unroll
    for (int w = 0; w < 4; ++w) {
      const float ph = params[(layer * 4 + w) * 3 + 0];
      const float th = params[(layer * 4 + w) * 3 + 1];
      const float om = params[(layer * 4 + w) * 3 + 2];
      const float ct = __cosf(th * 0.5f), st = __sinf(th * 0.5f);
      const float ap = 0.5f * (ph + om), am = 0.5f * (ph - om);
      const float cap = __cosf(ap), sap = __sinf(ap);
      const float cam = __cosf(am), sam = __sinf(am);
      rr[w][0][0] =  cap * ct;  ri[w][0][0] = -sap * ct;
      rr[w][0][1] = -cam * st;  ri[w][0][1] = -sam * st;
      rr[w][1][0] =  cam * st;  ri[w][1][0] = -sam * st;
      rr[w][1][1] =  cap * ct;  ri[w][1][1] =  sap * ct;
    }
    float ar = 0.f, ai = 0.f;
    for (int k = 0; k < 16; ++k) {
      float lr = 1.f, li = 0.f;
      #pragma unroll
      for (int w = 0; w < 4; ++w) {
        const int bi = (i >> (3 - w)) & 1, bk = (k >> (3 - w)) & 1;
        const float nr = lr * rr[w][bi][bk] - li * ri[w][bi][bk];
        const float ni = lr * ri[w][bi][bk] + li * rr[w][bi][bk];
        lr = nr; li = ni;
      }
      ar += lr * Ur[k][j] - li * Ui[k][j];
      ai += lr * Ui[k][j] + li * Ur[k][j];
    }
    __syncthreads();
    Ur[i][j] = ar; Ui[i][j] = ai;
    __syncthreads();
    const int r = (layer % 3) + 1;
    for (int w = 0; w < 4; ++w) {
      const int t = (w + r) & 3;
      const int src = ((i >> (3 - w)) & 1) ? (i ^ (1 << (3 - t))) : i;
      const float tr = Ur[src][j], ti = Ui[src][j];
      __syncthreads();
      Ur[i][j] = tr; Ui[i][j] = ti;
      __syncthreads();
    }
  }
  {
    const int p = __popc(j) & 3;
    const float pr = (p == 0) ? 1.f : (p == 2) ? -1.f : 0.f;
    const float pi = (p == 1) ? -1.f : (p == 3) ? 1.f : 0.f;
    const float ar = Ur[i][j] * pr - Ui[i][j] * pi;
    const float ai = Ur[i][j] * pi + Ui[i][j] * pr;
    __syncthreads();
    Ur[i][j] = ar; Ui[i][j] = ai;
    __syncthreads();
  }
  #pragma unroll
  for (int w = 0; w < 4; ++w) {
    float acc = 0.f;
    for (int q = 0; q < 16; ++q) {
      const float z = 1.f - 2.f * (float)((q >> (3 - w)) & 1);
      acc += z * (Ur[q][i] * Ur[q][j] + Ui[q][i] * Ui[q][j]);
    }
    wsM[(w * 16 + i) * 16 + j] = acc;
  }
}

// ---------------------------------------------------------------------------
// Kernel A2: one-time W2 (f32, [512][2048]) -> f16 copy in workspace.
// ---------------------------------------------------------------------------
__global__ void qff_cvt_w2(const float* __restrict__ W2,
                           _Float16* __restrict__ W2h, int n) {
  const int idx = (blockIdx.x * blockDim.x + threadIdx.x) * 4;
  if (idx < n) {
    const float4 v = *(const float4*)(W2 + idx);
    v4h h;
    h[0] = (_Float16)v.x; h[1] = (_Float16)v.y;
    h[2] = (_Float16)v.z; h[3] = (_Float16)v.w;
    *(v4h*)(W2h + idx) = h;
  }
}

// ---------------------------------------------------------------------------
// Kernel B: fused quantum-embed + FFN.
// 256 threads (8 wave32), 64 tokens/block. Waves: m_tile=wave&3 (16 tokens),
// n_blk=wave>>2 (256-wide N slice -> 16 C tiles, 128 acc VGPRs).
// Per 32-K chunk: W2 tile (512x32 f16) DMA'd to LDS with
// global_load_async_to_lds_b128 (double buffered, ASYNCcnt), h tile computed
// into LDS in WMMA A-fragment layout, 16x v_wmma_f32_16x16x32_f16 per wave.
// ---------------------------------------------------------------------------
__global__ __launch_bounds__(256)
void qff_main(const float* __restrict__ x,
              const float* __restrict__ W1,
              const float* __restrict__ b1,
              const _Float16* __restrict__ W2h,
              const float* __restrict__ b2,
              const float* __restrict__ wsM,
              float* __restrict__ out) {
  __shared__ float    Ms[4][16][16];
  __shared__ float    ez[QFF_TOK][4];
  __shared__ _Float16 hA[QFF_TOK][HA_PITCH];
  __shared__ _Float16 Bs[2][QFF_EMBED][HA_PITCH];

  const int  tid  = threadIdx.x;
  const long tok0 = (long)blockIdx.x * QFF_TOK;

  // Issue async DMA of W2 chunk 0 into buffer 0 (overlaps expz phase).
  #pragma unroll
  for (int t = 0; t < 8; ++t) {
    const int L = tid + t * 256;             // 0..2047 b128 segments
    const int n = L >> 2, seg = (L & 3) * 8; // row n, 8-half segment
    async_b128(W2h + (long)n * QFF_FFN + seg, &Bs[0][n][seg]);
  }

  // Stage M matrices into LDS.
  #pragma unroll
  for (int it = 0; it < 4; ++it) {
    const int lin = tid + it * 256;
    ((float*)Ms)[lin] = wsM[lin];
  }
  __syncthreads();

  // expz: 4 threads per token, one quadratic form each (all 256 threads).
  {
    const int tl = tid >> 2, w = tid & 3;
    const float4 ang = *(const float4*)(x + (tok0 + tl) * QFF_EMBED);
    const float c0 = __cosf(ang.x * 0.5f), s0 = __sinf(ang.x * 0.5f);
    const float c1 = __cosf(ang.y * 0.5f), s1 = __sinf(ang.y * 0.5f);
    const float c2 = __cosf(ang.z * 0.5f), s2 = __sinf(ang.z * 0.5f);
    const float c3 = __cosf(ang.w * 0.5f), s3 = __sinf(ang.w * 0.5f);
    float m[16];
    #pragma unroll
    for (int jj = 0; jj < 16; ++jj) {
      float v = ((jj >> 3) & 1) ? s0 : c0;
      v *= ((jj >> 2) & 1) ? s1 : c1;
      v *= ((jj >> 1) & 1) ? s2 : c2;
      v *= ( jj       & 1) ? s3 : c3;
      m[jj] = v;
    }
    float acc = 0.f;
    #pragma unroll
    for (int jj = 0; jj < 16; ++jj) {
      float rd = 0.f;
      #pragma unroll
      for (int kk = 0; kk < 16; ++kk) rd += Ms[w][jj][kk] * m[kk];
      acc += m[jj] * rd;
    }
    ez[tl][w] = acc;
  }
  __syncthreads();

  const int lane     = tid & 31;
  const int wave     = tid >> 5;
  const int m_tile   = wave & 3;    // 0..3 : 16-token rows
  const int n_blk    = wave >> 2;   // 0..1 : 256-wide N slice
  const int lane_lo  = lane & 15;
  const int half_sel = lane >> 4;

  v8f acc[16] = {};

  // h producer: each thread makes 8 h values per chunk (64 tok x 32 k).
  const int   tl_h = tid >> 2, kg = tid & 3;
  const float e0 = ez[tl_h][0], e1 = ez[tl_h][1];
  const float e2 = ez[tl_h][2], e3 = ez[tl_h][3];

  // A fragment: 16-bit 16x32 layout; lane<16: K {0..7,16..23}, lane>=16:
  // K {8..15,24..31} of row M=lane%16.
  const _Float16* Arow = &hA[m_tile * 16 + lane_lo][half_sel * 8];

  for (int c = 0; c < QFF_FFN / KCHUNK; ++c) {
    const int kc  = c * KCHUNK;
    const int buf = c & 1;

    // DMA next W2 chunk into the other buffer.
    if (c + 1 < QFF_FFN / KCHUNK) {
      #pragma unroll
      for (int t = 0; t < 8; ++t) {
        const int L = tid + t * 256;
        const int n = L >> 2, seg = (L & 3) * 8;
        async_b128(W2h + (long)n * QFF_FFN + (kc + KCHUNK) + seg,
                   &Bs[buf ^ 1][n][seg]);
      }
    }

    // h chunk: relu(W1 @ expz + b1) -> f16 LDS tile.
    {
      const int k0 = kc + kg * 8;
      const float4 bb0 = *(const float4*)(b1 + k0);
      const float4 bb1 = *(const float4*)(b1 + k0 + 4);
      const float bias[8] = {bb0.x, bb0.y, bb0.z, bb0.w,
                             bb1.x, bb1.y, bb1.z, bb1.w};
      v8h hh;
      #pragma unroll
      for (int d = 0; d < 8; ++d) {
        const float4 w = *(const float4*)(W1 + (long)(k0 + d) * 4);
        const float v = w.x * e0 + w.y * e1 + w.z * e2 + w.w * e3 + bias[d];
        hh[d] = (_Float16)fmaxf(v, 0.f);
      }
      *(v8h*)(&hA[tl_h][kg * 8]) = hh;
    }

    // Wait for this chunk's DMA (in-order completion), then make visible.
    if (c + 1 < QFF_FFN / KCHUNK) { ASYNC_WAIT(8); } else { ASYNC_WAIT(0); }
    __syncthreads();

    // A fragment (two ds_read_b128).
    const v8h alo = *(const v8h*)(Arow);
    const v8h ahi = *(const v8h*)(Arow + 16);
    v16h a;
    #pragma unroll
    for (int e = 0; e < 8; ++e) { a[e] = alo[e]; a[e + 8] = ahi[e]; }

    // B fragments from LDS: lane<16: N=lane, K {0..7,16..23}; lane>=16: K {8..15,24..31}.
    #pragma unroll
    for (int nt = 0; nt < 16; ++nt) {
      const _Float16* bp =
          &Bs[buf][n_blk * 256 + nt * 16 + lane_lo][half_sel * 8];
      const v8h blo = *(const v8h*)(bp);
      const v8h bhi = *(const v8h*)(bp + 16);
      v16h b;
      #pragma unroll
      for (int e = 0; e < 8; ++e) { b[e] = blo[e]; b[e + 8] = bhi[e]; }
      acc[nt] = __builtin_amdgcn_wmma_f32_16x16x32_f16(
          false, a, false, b, (short)0, acc[nt], false, false);
    }
    __syncthreads();
  }

  // Epilogue: C layout lane<16: N=lane, M=r; lane>=16: N=lane-16, M=8+r.
  #pragma unroll
  for (int nt = 0; nt < 16; ++nt) {
    const int   n16  = n_blk * 256 + nt * 16 + lane_lo;
    const float bias = b2[n16];
    #pragma unroll
    for (int r = 0; r < 8; ++r) {
      const int mrow = half_sel * 8 + r;
      out[(tok0 + m_tile * 16 + mrow) * QFF_EMBED + n16] = acc[nt][r] + bias;
    }
  }
}

// ---------------------------------------------------------------------------
extern "C" void kernel_launch(void* const* d_in, const int* in_sizes, int n_in,
                              void* d_out, int out_size, void* d_ws,
                              size_t ws_size, hipStream_t stream) {
  const float* x      = (const float*)d_in[0];  // (8, 8192, 512) f32
  const float* params = (const float*)d_in[1];  // (2, 4, 3) f32
  const float* W1     = (const float*)d_in[2];  // (2048, 4) f32
  const float* b1     = (const float*)d_in[3];  // (2048,) f32
  const float* W2     = (const float*)d_in[4];  // (512, 2048) f32
  const float* b2     = (const float*)d_in[5];  // (512,) f32
  float*       out    = (float*)d_out;          // (8, 8192, 512) f32

  float*    wsM = (float*)d_ws;                      // 4 KB: M matrices
  _Float16* W2h = (_Float16*)((char*)d_ws + 4096);   // 2 MB: f16 W2

  const int tokens = in_sizes[0] / QFF_EMBED;        // 65536
  const int nW2    = in_sizes[4];                    // 1048576

  qff_build_M<<<1, 256, 0, stream>>>(params, wsM);
  qff_cvt_w2<<<(nW2 / 4 + 255) / 256, 256, 0, stream>>>(W2, W2h, nW2);
  qff_main<<<tokens / QFF_TOK, 256, 0, stream>>>(x, W1, b1, W2h, b2, wsM, out);
}
